// SoftplusPWP_40973988004762
// MI455X (gfx1250) — matrix operand order repaired and checked by reading, not compiled, not run on hardware
//
#include <hip/hip_runtime.h>
#include <hip/hip_bf16.h>
#include <math.h>

typedef __attribute__((ext_vector_type(16))) _Float16 v16h;
typedef __attribute__((ext_vector_type(8)))  float    v8f;

#define BATCH 2
#define SEQ   1024
#define DIM   1024
#define HEADS 16
#define DHEAD 64

union H2 { unsigned u; _Float16 h[2]; };
union H4 { uint2 u; _Float16 h[4]; };
union H8 { uint4 u; _Float16 h[8]; };

// ---------------------------------------------------------------------------
// CDNA5 async copy: GLOBAL_LOAD_ASYNC_TO_LDS_B128 (per-lane 16B, ASYNCcnt).
// GV mode: vdst = LDS byte address, vaddr = 64-bit global address.
// ---------------------------------------------------------------------------
__device__ __forceinline__ void async_ld16(uint32_t lds_addr, uint64_t gaddr) {
    asm volatile("global_load_async_to_lds_b128 %0, %1, off"
                 :: "v"(lds_addr), "v"(gaddr) : "memory");
}
__device__ __forceinline__ uint32_t lds_lo32(const void* p) {
    return (uint32_t)(uintptr_t)p;   // generic LDS addr: low 32 bits = LDS offset
}
__device__ __forceinline__ void wait_async0() {
#if __has_builtin(__builtin_amdgcn_s_wait_asynccnt)
    __builtin_amdgcn_s_wait_asynccnt(0);
#else
    asm volatile("s_wait_asynccnt 0x0" ::: "memory");
#endif
}

// ---------------------------------------------------------------------------
// Kernel 1: per-row LayerNorm -> xn16 (fp16), plus q/k projections
// (after the reference's reshapes, q/k are one scalar per (b,h,l)).
// ---------------------------------------------------------------------------
__global__ __launch_bounds__(256) void ln_qk_kernel(
    const float* __restrict__ x,
    const float* __restrict__ Wq, const float* __restrict__ bq,
    const float* __restrict__ Wk, const float* __restrict__ bk,
    const float* __restrict__ gam, const float* __restrict__ bet,
    _Float16* __restrict__ xn16,
    float* __restrict__ qf, float* __restrict__ kf)
{
    __shared__ float s_x[DIM];
    __shared__ float red0[256];
    __shared__ float red1[256];
    int tid = threadIdx.x;
    int row = blockIdx.x;
    int b = row / SEQ, l = row % SEQ;

    const float4* xr = (const float4*)(x + (size_t)row * DIM);
    float4 xv = xr[tid];
    float s  = xv.x + xv.y + xv.z + xv.w;
    float ss = xv.x*xv.x + xv.y*xv.y + xv.z*xv.z + xv.w*xv.w;
    red0[tid] = s; red1[tid] = ss;
    __syncthreads();
    for (int off = 128; off > 0; off >>= 1) {
        if (tid < off) { red0[tid] += red0[tid+off]; red1[tid] += red1[tid+off]; }
        __syncthreads();
    }
    float mu   = red0[0] * (1.0f / DIM);
    float var  = red1[0] * (1.0f / DIM) - mu * mu;
    float rstd = rsqrtf(var + 1e-5f);
    __syncthreads();

    int i0 = tid * 4;
    H4 pk4;
    const float* xs = (const float*)&xv;
    for (int u = 0; u < 4; ++u) {
        float v = (xs[u] - mu) * rstd * gam[i0+u] + bet[i0+u];
        s_x[i0+u] = v;
        pk4.h[u] = (_Float16)v;
    }
    ((uint2*)(xn16 + (size_t)row * DIM))[tid] = pk4.u;
    __syncthreads();

    int c = tid & 15, grp = tid >> 4;
    float pq = 0.f, pk = 0.f;
    for (int i = grp*64; i < grp*64 + 64; ++i) {
        float v = s_x[i];
        pq += v * Wq[i*16 + c];
        pk += v * Wk[i*16 + c];
    }
    red0[tid] = pq; red1[tid] = pk;
    __syncthreads();
    for (int off = 128; off >= 16; off >>= 1) {
        if (tid < off) { red0[tid] += red0[tid+off]; red1[tid] += red1[tid+off]; }
        __syncthreads();
    }
    if (tid < 16) {
        qf[((size_t)b*HEADS + tid)*SEQ + l] = red0[tid] + bq[tid];
        kf[((size_t)b*HEADS + tid)*SEQ + l] = red1[tid] + bk[tid];
    }
}

// ---------------------------------------------------------------------------
// Kernel 1b: Wv (f32, [k][n]) -> WvT (fp16, [n][k]) via LDS tile transpose,
// so the GEMM B-tile becomes a straight async 16B-chunk copy.
// ---------------------------------------------------------------------------
__global__ __launch_bounds__(256) void wv_transpose_kernel(
    const float* __restrict__ Wv, _Float16* __restrict__ WvT)
{
    __shared__ _Float16 t[64 * 72];   // [n][k], stride 72 to dodge bank conflicts
    int tid = threadIdx.x;
    int k0 = blockIdx.x * 64;
    int n0 = blockIdx.y * 64;

    int kk  = tid >> 2;
    int nn0 = (tid & 3) * 16;
    const float4* src = (const float4*)(Wv + (size_t)(k0 + kk) * DIM + n0 + nn0);
    for (int q = 0; q < 4; ++q) {
        float4 f = src[q];
        t[(nn0 + q*4 + 0)*72 + kk] = (_Float16)f.x;
        t[(nn0 + q*4 + 1)*72 + kk] = (_Float16)f.y;
        t[(nn0 + q*4 + 2)*72 + kk] = (_Float16)f.z;
        t[(nn0 + q*4 + 3)*72 + kk] = (_Float16)f.w;
    }
    __syncthreads();

    int nn = tid >> 2;
    int cc = (tid & 3) * 16;
    H8 a, b;
    for (int i = 0; i < 8; ++i) { a.h[i] = t[nn*72 + cc + i]; b.h[i] = t[nn*72 + cc + 8 + i]; }
    uint4* dst = (uint4*)(WvT + (size_t)(n0 + nn) * DIM + k0 + cc);
    dst[0] = a.u;
    dst[1] = b.u;
}

// ---------------------------------------------------------------------------
// Kernel 2: v = xn @ Wv + bv, fp16 WMMA, f32 acc, ASYNC double-buffered LDS.
// Output written TRANSPOSED as vT16[(b*DIM + d)][j] (fp16) so kernel 3's
// B tile is also a straight async copy; the C fragment's 8 consecutive M
// values per lane make this a single b128 store per subtile.
// ---------------------------------------------------------------------------
__global__ __launch_bounds__(256) void gemm_v_kernel(
    const _Float16* __restrict__ xn16,
    const _Float16* __restrict__ WvT,
    const float* __restrict__ bv,
    _Float16* __restrict__ vT16)
{
    __shared__ _Float16 aT[2][128 * 32];   // [m][k]
    __shared__ _Float16 bT[2][64 * 32];    // [n][k]
    int tid  = threadIdx.x;
    int lane = tid & 31;
    int wave = tid >> 5;
    int hi   = (lane >> 4) & 1;
    int mBase = blockIdx.x * 128;
    int nBase = blockIdx.y * 64;

    auto stage = [&](int buf, int k0) {
        int i    = tid >> 1;
        int koff = (tid & 1) * 16;
        uint64_t ga = (uint64_t)(uintptr_t)(xn16 + (size_t)(mBase + i) * DIM + k0 + koff);
        uint32_t la = lds_lo32(&aT[buf][i*32 + koff]);
        async_ld16(la,      ga);
        async_ld16(la + 16, ga + 16);
        int n  = tid >> 2;
        int c8 = (tid & 3) * 8;
        uint64_t gb = (uint64_t)(uintptr_t)(WvT + (size_t)(nBase + n) * DIM + k0 + c8);
        async_ld16(lds_lo32(&bT[buf][n*32 + c8]), gb);
    };

    v8f acc[4] = {};
    stage(0, 0);
    wait_async0();
    __syncthreads();

    for (int k0 = 0; k0 < DIM; k0 += 32) {
        int buf = (k0 >> 5) & 1;
        if (k0 + 32 < DIM) stage(buf ^ 1, k0 + 32);

        v16h afrag;
        int ml = wave*16 + (lane & 15);
        for (int r = 0; r < 8; ++r) {
            int kp = ((r < 4) ? (2*r) : (16 + 2*(r-4))) + (hi ? 8 : 0);
            H2 p; p.u = *(const unsigned*)(&aT[buf][ml*32 + kp]);
            afrag[2*r] = p.h[0]; afrag[2*r+1] = p.h[1];
        }
        for (int t = 0; t < 4; ++t) {
            v16h bfrag;
            int n  = t*16 + (lane & 15);
            int kb = hi ? 8 : 0;
            for (int r = 0; r < 8; ++r) {
                H2 p; p.u = *(const unsigned*)(&bT[buf][n*32 + kb + ((r < 4) ? (2*r) : (16 + 2*(r-4)))]);
                bfrag[2*r] = p.h[0]; bfrag[2*r+1] = p.h[1];
            }
            acc[t] = __builtin_amdgcn_wmma_f32_16x16x32_f16(
                false, afrag, false, bfrag, (short)0, acc[t], false, false);
        }
        wait_async0();
        __syncthreads();
    }

    for (int t = 0; t < 4; ++t) {
        int d = nBase + t*16 + (lane & 15);
        float bias = bv[d];
        int m0 = mBase + wave*16 + hi*8;
        int b  = m0 >> 10;           // SEQ = 1024
        int j0 = m0 & (SEQ - 1);
        H8 p;
        for (int r = 0; r < 8; ++r) p.h[r] = (_Float16)(acc[t][r] + bias);
        *(uint4*)(vT16 + ((size_t)(b*DIM + d))*SEQ + j0) = p.u;
    }
}

// ---------------------------------------------------------------------------
// Kernel 3: scores computed on-the-fly in WMMA A-layout, WMMA against the
// async double-buffered v tile, gelu, residual.
// ---------------------------------------------------------------------------
__global__ __launch_bounds__(256) void attn_kernel(
    const float* __restrict__ x,
    const float* __restrict__ mask,
    const _Float16* __restrict__ vT16,
    const float* __restrict__ qf,
    const float* __restrict__ kf,
    const float* __restrict__ zf_p,
    const float* __restrict__ angle,
    float* __restrict__ out)
{
    __shared__ _Float16 vL[2][DHEAD * 32];   // [n][j]
    __shared__ float s_k[2][32];
    __shared__ float s_m[2][32];

    int tid  = threadIdx.x;
    int lane = tid & 31;
    int wave = tid >> 5;
    int hi   = (lane >> 4) & 1;

    int bh = blockIdx.x;
    int b = bh >> 4;
    int h = bh & 15;
    int rowBase = blockIdx.y * 128 + wave * 16;

    float ang = angle[h];
    float ca = cosf(ang), sa = sinf(ang);
    float scale = sqrtf(2.0f / (float)SEQ);
    float zf = zf_p[0];

    float qv = qf[((size_t)b*HEADS + h)*SEQ + rowBase + (lane & 15)];

    auto stageV = [&](int buf, int j0) {
        int n  = tid >> 2;
        int c8 = (tid & 3) * 8;
        uint64_t g = (uint64_t)(uintptr_t)(vT16 + ((size_t)(b*DIM) + h*DHEAD + n)*SEQ + j0 + c8);
        async_ld16(lds_lo32(&vL[buf][n*32 + c8]), g);
    };
    auto stageKM = [&](int buf, int j0) {
        if (tid < 32)      s_k[buf][tid]      = kf[((size_t)b*HEADS + h)*SEQ + j0 + tid];
        else if (tid < 64) s_m[buf][tid - 32] = mask[(size_t)b*SEQ + j0 + (tid - 32)];
    };

    v8f acc[4] = {};
    stageV(0, 0);
    stageKM(0, 0);
    wait_async0();
    __syncthreads();

    for (int j0 = 0; j0 < SEQ; j0 += 32) {
        int buf = (j0 >> 5) & 1;
        if (j0 + 32 < SEQ) { stageV(buf ^ 1, j0 + 32); stageKM(buf ^ 1, j0 + 32); }

        v16h afrag;
        for (int r = 0; r < 8; ++r) {
            int kp = ((r < 4) ? (2*r) : (16 + 2*(r-4))) + (hi ? 8 : 0);
            for (int u = 0; u < 2; ++u) {
                int kk = kp + u;
                float kv   = s_k[buf][kk];
                float real = ca*qv - sa*kv;
                float imag = sa*qv + ca*kv;
                float mx   = fmaxf(real, imag);
                float er   = __expf(real - mx);
                float ei   = __expf(imag - mx);
                float comp = (er*real + ei*imag) / (er + ei);
                float sp   = fmaxf(comp, 0.0f) + log1pf(__expf(-fabsf(comp)));
                sp *= (1.0f - s_m[buf][kk]) * scale;
                afrag[2*r + u] = (_Float16)sp;
            }
        }

        for (int t = 0; t < 4; ++t) {
            v16h bfrag;
            int n  = t*16 + (lane & 15);
            int kb = hi ? 8 : 0;
            for (int r = 0; r < 8; ++r) {
                H2 p; p.u = *(const unsigned*)(&vL[buf][n*32 + kb + ((r < 4) ? (2*r) : (16 + 2*(r-4)))]);
                bfrag[2*r] = p.h[0]; bfrag[2*r+1] = p.h[1];
            }
            acc[t] = __builtin_amdgcn_wmma_f32_16x16x32_f16(
                false, afrag, false, bfrag, (short)0, acc[t], false, false);
        }
        wait_async0();
        __syncthreads();
    }

    const float inv_sqrt2 = 0.70710678118654752f;
    for (int t = 0; t < 4; ++t) {
        int d = h*DHEAD + t*16 + (lane & 15);
        for (int r = 0; r < 8; ++r) {
            int l = rowBase + r + (hi ? 8 : 0);
            float val = acc[t][r];
            float g = 0.5f * val * (1.0f + erff(val * inv_sqrt2));
            size_t idx = ((size_t)b*SEQ + l)*DIM + d;
            out[idx] = x[idx] + g * zf;
        }
    }
}

// ---------------------------------------------------------------------------
extern "C" void kernel_launch(void* const* d_in, const int* in_sizes, int n_in,
                              void* d_out, int out_size, void* d_ws, size_t ws_size,
                              hipStream_t stream) {
    const float* x     = (const float*)d_in[0];
    const float* mask  = (const float*)d_in[1];
    const float* Wq    = (const float*)d_in[2];
    const float* bq    = (const float*)d_in[3];
    const float* Wk    = (const float*)d_in[4];
    const float* bk    = (const float*)d_in[5];
    const float* Wv    = (const float*)d_in[6];
    const float* bv    = (const float*)d_in[7];
    const float* ln_g  = (const float*)d_in[8];
    const float* ln_b  = (const float*)d_in[9];
    const float* zf    = (const float*)d_in[10];
    const float* angle = (const float*)d_in[11];
    float* out = (float*)d_out;

    char* ws = (char*)d_ws;
    _Float16* xn16 = (_Float16*)ws;                                 // 4 MB
    _Float16* vT16 = (_Float16*)(ws + (size_t)4*1024*1024);         // 4 MB
    _Float16* WvT  = (_Float16*)(ws + (size_t)8*1024*1024);         // 2 MB
    float*    qf   = (float*)(ws + (size_t)10*1024*1024);           // 128 KB
    float*    kf   = (float*)(ws + (size_t)10*1024*1024 + 131072);  // 128 KB

    ln_qk_kernel<<<BATCH*SEQ, 256, 0, stream>>>(x, Wq, bq, Wk, bk, ln_g, ln_b,
                                                xn16, qf, kf);
    wv_transpose_kernel<<<dim3(DIM/64, DIM/64), 256, 0, stream>>>(Wv, WvT);
    gemm_v_kernel<<<dim3((BATCH*SEQ)/128, DIM/64), 256, 0, stream>>>(xn16, WvT, bv, vT16);
    attn_kernel<<<dim3(BATCH*HEADS, SEQ/128), 256, 0, stream>>>(x, mask, vT16,
                                                                qf, kf, zf, angle, out);
}
